// SEBlock_11879879543965
// MI455X (gfx1250) — matrix operand order repaired and checked by reading, not compiled, not run on hardware
//
#include <hip/hip_runtime.h>
#include <math.h>

typedef __attribute__((ext_vector_type(16))) _Float16 v16h;
typedef __attribute__((ext_vector_type(8)))  float    v8f;
typedef __attribute__((ext_vector_type(4)))  float    f32x4;

static constexpr int kB   = 32;    // batch
static constexpr int kC   = 512;   // channels
static constexpr int kHid = 32;    // hidden = C/16
static constexpr int kHW  = 3136;  // 56*56
static constexpr int kHW4 = 784;   // 3136/4

// ---------------------------------------------------------------------------
// Pass 1: per-(b,c) plane mean.  One block per plane, 256 threads, float4 loads.
// RT loads deliberately fill L2 (192 MB) so pass 3 re-reads mostly hit L2.
// ---------------------------------------------------------------------------
__global__ void se_squeeze(const float* __restrict__ x, float* __restrict__ mean) {
    const int plane = blockIdx.x;                       // 0 .. B*C-1
    const f32x4* xv = (const f32x4*)x + (size_t)plane * kHW4;
    float s = 0.0f;
    for (int i = threadIdx.x; i < kHW4; i += 256) {
        f32x4 v = xv[i];
        s += v.x + v.y + v.z + v.w;
    }
    // wave32 butterfly reduction
    for (int off = 16; off > 0; off >>= 1)
        s += __shfl_xor(s, off, 32);
    __shared__ float psum[8];
    const int lane = threadIdx.x & 31;
    const int wave = threadIdx.x >> 5;
    if (lane == 0) psum[wave] = s;
    __syncthreads();
    if (threadIdx.x == 0) {
        float t = 0.0f;
        for (int w = 0; w < 8; ++w) t += psum[w];
        mean[plane] = t * (1.0f / (float)kHW);
    }
}

// ---------------------------------------------------------------------------
// Pass 2: excite.  One workgroup (8 waves).  Both GEMMs use
// v_wmma_f32_16x16x32_f16.  ISA 7.12.2 per-lane layouts:
//   A 16x32 f16 : row = lane&15 ; half j -> K = (j&7) + 16*(j>>3) + 8*(lane>>4)
//   B 32x16 f16 : col = lane&15 ; half j -> K = (j&7) + 16*(j>>3) + 8*(lane>>4)
//   C/D 16x16 f32: VGPR r -> M = r + 8*(lane>>4), N = lane&15
// All waves run the WMMAs (EXEC must be all 1s); stores are predicated.
// ---------------------------------------------------------------------------
__global__ void se_excite(const float* __restrict__ mean,
                          const float* __restrict__ w1, const float* __restrict__ b1,
                          const float* __restrict__ w2, const float* __restrict__ b2,
                          const float* __restrict__ prelu_w,
                          float* __restrict__ yscale) {
    const int tid   = threadIdx.x;
    const int wave  = tid >> 5;
    const int lane  = tid & 31;
    const int lhalf = lane >> 4;       // 0 or 1
    const int l16   = lane & 15;

    __shared__ _Float16 a2[32][32];    // PReLU(fc1) intermediate, f16

    // ---- GEMM1: Z1(32x32) = mean(32x512) @ W1^T  (B[k][n] = w1[n*512+k])
    {
        const int tile = wave & 3;     // waves 4-7 duplicate 0-3 (keeps EXEC full)
        const int mt = tile & 1, nt = tile >> 1;
        v8f acc = {};
        for (int kk = 0; kk < 16; ++kk) {
            const int kbase = kk * 32;
            v16h a, bm;
            for (int j = 0; j < 16; ++j) {
                const int kA = (j & 7) + ((j >> 3) << 4) + (lhalf << 3);
                const int kBi = (j & 7) + ((j >> 3) << 4) + (lhalf << 3);
                a[j]  = (_Float16)mean[(mt * 16 + l16) * kC + kbase + kA];
                bm[j] = (_Float16)w1[(nt * 16 + l16) * kC + kbase + kBi];
            }
            acc = __builtin_amdgcn_wmma_f32_16x16x32_f16(
                false, a, false, bm, (short)0, acc, false, false);
        }
        if (wave < 4) {
            const float pw   = prelu_w[0];
            const int   h    = nt * 16 + l16;
            const float bias = b1[h];
            for (int r = 0; r < 8; ++r) {
                const int brow = mt * 16 + r + (lhalf << 3);
                float z = acc[r] + bias;
                z = (z > 0.0f) ? z : pw * z;           // PReLU
                a2[brow][h] = (_Float16)z;
            }
        }
    }
    __syncthreads();

    // ---- GEMM2: Z2(32x512) = A2(32x32) @ W2^T  (B[k][n] = w2[n*32+k])
    for (int t = wave; t < 64; t += 8) {               // 8 tiles per wave, uniform
        const int mt = t & 1, nt = t >> 1;
        v16h a, bm;
        for (int j = 0; j < 16; ++j) {
            const int kA  = (j & 7) + ((j >> 3) << 4) + (lhalf << 3);
            const int kBi = (j & 7) + ((j >> 3) << 4) + (lhalf << 3);
            a[j]  = a2[mt * 16 + l16][kA];
            bm[j] = (_Float16)w2[(nt * 16 + l16) * kHid + kBi];
        }
        v8f acc = {};
        acc = __builtin_amdgcn_wmma_f32_16x16x32_f16(
            false, a, false, bm, (short)0, acc, false, false);
        const int   ch   = nt * 16 + l16;
        const float bias = b2[ch];
        for (int r = 0; r < 8; ++r) {
            const int brow = mt * 16 + r + (lhalf << 3);
            const float v  = acc[r] + bias;
            yscale[brow * kC + ch] = 1.0f / (1.0f + expf(-v));   // sigmoid
        }
    }
}

// ---------------------------------------------------------------------------
// Pass 3: out = x * y[b,c].  One block per plane.  NT loads (last use of x,
// should hit L2 from pass 1) and NT stores (streaming, don't thrash L2).
// ---------------------------------------------------------------------------
__global__ void se_scale(const float* __restrict__ x,
                         const float* __restrict__ yscale,
                         float* __restrict__ out) {
    const int plane = blockIdx.x;
    const float s = yscale[plane];
    const f32x4* xv = (const f32x4*)x   + (size_t)plane * kHW4;
    f32x4*       ov = (f32x4*)out       + (size_t)plane * kHW4;
    for (int i = threadIdx.x; i < kHW4; i += 256) {
        f32x4 v = __builtin_nontemporal_load(&xv[i]);
        v = v * s;
        __builtin_nontemporal_store(v, &ov[i]);
    }
}

// ---------------------------------------------------------------------------
extern "C" void kernel_launch(void* const* d_in, const int* in_sizes, int n_in,
                              void* d_out, int out_size, void* d_ws, size_t ws_size,
                              hipStream_t stream) {
    const float* x  = (const float*)d_in[0];
    const float* w1 = (const float*)d_in[1];
    const float* b1 = (const float*)d_in[2];
    const float* w2 = (const float*)d_in[3];
    const float* b2 = (const float*)d_in[4];
    const float* pw = (const float*)d_in[5];
    float* out = (float*)d_out;

    float* mean   = (float*)d_ws;            // 32*512 floats
    float* yscale = mean + kB * kC;          // 32*512 floats  (128 KB total ws use)

    se_squeeze<<<kB * kC, 256, 0, stream>>>(x, mean);
    se_excite<<<1, 256, 0, stream>>>(mean, w1, b1, w2, b2, pw, yscale);
    se_scale<<<kB * kC, 256, 0, stream>>>(x, yscale, out);
}
